// TransformerClassifier_44933947851188
// MI455X (gfx1250) — matrix operand order repaired and checked by reading, not compiled, not run on hardware
//
#include <hip/hip_runtime.h>

typedef __attribute__((ext_vector_type(16))) __bf16 v16bf;
typedef __attribute__((ext_vector_type(8)))  __bf16 v8bf;
typedef __attribute__((ext_vector_type(8)))  float  v8f;

namespace {
constexpr int kB  = 16384;   // batch
constexpr int kD  = 768;     // dim
constexpr int kA  = 51;      // attributes
constexpr int kAP = 64;      // attributes padded to WMMA N multiple
constexpr float kEps = 1e-5f;

__device__ __forceinline__ v16bf cat8(v8bf lo, v8bf hi) {
  return __builtin_shufflevector(lo, hi, 0, 1, 2, 3, 4, 5, 6, 7,
                                          8, 9, 10, 11, 12, 13, 14, 15);
}

// One per-lane 16-byte async global->LDS copy (ASYNCcnt-tracked DMA).
__device__ __forceinline__ void async_copy16(uint32_t lds_byte_off,
                                             const void* gsrc) {
  asm volatile("global_load_async_to_lds_b128 %0, %1, off"
               :: "v"(lds_byte_off), "v"((uint64_t)(uintptr_t)gsrc)
               : "memory");
}

__device__ __forceinline__ void wait_async0() {
  asm volatile("s_wait_asynccnt 0" ::: "memory");
}

__device__ __forceinline__ v8f wmma_bf16(v16bf a, v16bf b, v8f c) {
  return __builtin_amdgcn_wmma_f32_16x16x32_bf16(false, a, false, b,
                                                 (short)0, c, false, false);
}
} // namespace

// ---- conversion / setup kernels -------------------------------------------

__global__ void k_cvt_bf16(const float* __restrict__ src,
                           __bf16* __restrict__ dst, int n) {
  int i = blockIdx.x * blockDim.x + threadIdx.x;
  if (i < n) dst[i] = (__bf16)src[i];
}

__global__ void k_pad_wattr(const float* __restrict__ w,   // [51,768]
                            __bf16* __restrict__ dst) {    // [64,768]
  int i = blockIdx.x * blockDim.x + threadIdx.x;
  if (i >= kAP * kD) return;
  int r = i / kD;
  dst[i] = (r < kA) ? (__bf16)w[i] : (__bf16)0.f;
}

__global__ void k_zero_f32(float* __restrict__ p, int n) {
  int i = blockIdx.x * blockDim.x + threadIdx.x;
  if (i < n) p[i] = 0.f;
}

// ---- GEMM1: h = x @ W_embed^T + b_embed  (bf16 in, bf16 out, f32 accum) ----
// grid: (M/256, D/64), block: 256 (8 waves). Wave w owns rows [w*32, w*32+32)
// (two 16-row WMMA subtiles -> 8 accumulators), across the 64-col tile.
// B tile (64 x 64K bf16 = 8 KB) staged in LDS by async DMA (2 chunks/thread),
// double-buffered; each stage feeds 2 compute sub-steps of 8 WMMAs.

__global__ __launch_bounds__(256) void k_gemm_embed(
    const __bf16* __restrict__ Xb,   // [16384,768] row-major
    const __bf16* __restrict__ Wb,   // [768,768] row-major (rows = output cols)
    const float*  __restrict__ bias, // [768]
    __bf16* __restrict__ Hb) {       // [16384,768]
  __shared__ __align__(16) __bf16 sB[2][64 * 64];   // 2 x 8 KB double buffer

  const int tid  = threadIdx.x;
  const int lane = tid & 31;
  const int wave = tid >> 5;
  const int l15  = lane & 15;
  const int hi   = lane >> 4;                 // half-wave select
  const int row0 = blockIdx.x * 256 + wave * 32;
  const int n0   = blockIdx.y * 64;

  // staging: 512 chunks of 16B per stage; thread copies rows crow, crow+32
  const int crow = tid >> 3;                  // 0..31
  const int csub = tid & 7;                   // 0..7 (16B chunk in 128B row)
  const __bf16* gB0 = Wb + (size_t)(n0 + crow) * kD + csub * 8;
  const __bf16* gB1 = gB0 + (size_t)32 * kD;
  // LDS aperture: flat address low 32 bits == LDS byte offset (ISA 10.2)
  const uint32_t ldsBase = (uint32_t)(uintptr_t)(&sB[0][0]);
  const uint32_t lds0 = ldsBase + (uint32_t)(crow * 128 + csub * 16);
  const uint32_t lds1 = lds0 + 32u * 128u;

  // prefetch stage 0
  async_copy16(lds0, gB0);
  async_copy16(lds1, gB1);

  // A fragment bases: lane = M row; lanes 16-31 hold K+8 / K+24 (ISA 7.12.2)
  const __bf16* aB0 = Xb + (size_t)(row0 + l15) * kD + hi * 8;
  const __bf16* aB1 = aB0 + (size_t)16 * kD;

  v8f acc[8] = {v8f{}, v8f{}, v8f{}, v8f{}, v8f{}, v8f{}, v8f{}, v8f{}};

  for (int k0 = 0; k0 < kD; k0 += 64) {
    const int buf = (k0 >> 6) & 1;
    wait_async0();          // my DMA done
    __syncthreads();        // everyone's DMA done -> tile visible
    if (k0 + 64 < kD) {     // overlap next stage's DMA with this stage's math
      const uint32_t bo = (uint32_t)((buf ^ 1) * 8192);
      async_copy16(lds0 + bo, gB0 + (k0 + 64));
      async_copy16(lds1 + bo, gB1 + (k0 + 64));
    }

    const __bf16* sb = &sB[buf][0];
#pragma unroll
    for (int s = 0; s < 2; ++s) {
      const int ke = k0 + s * 32;
      v16bf a0 = cat8(*(const v8bf*)(aB0 + ke), *(const v8bf*)(aB0 + ke + 16));
      v16bf a1 = cat8(*(const v8bf*)(aB1 + ke), *(const v8bf*)(aB1 + ke + 16));
      v16bf bfrag[4];
#pragma unroll
      for (int j = 0; j < 4; ++j) {
        // B frag: lane = N col; lanes 0-15 K..K+7,K+16..K+23, lanes 16-31 K+8..K+15,K+24..K+31
        const __bf16* bp = sb + (j * 16 + l15) * 64 + s * 32 + hi * 8;
        bfrag[j] = cat8(*(const v8bf*)bp, *(const v8bf*)(bp + 16));
      }
#pragma unroll
      for (int j = 0; j < 4; ++j) acc[j]     = wmma_bf16(a0, bfrag[j], acc[j]);
#pragma unroll
      for (int j = 0; j < 4; ++j) acc[4 + j] = wmma_bf16(a1, bfrag[j], acc[4 + j]);
    }
  }

  // C/D layout: VGPR r -> M = r (+8 for upper half-wave), N = lane&15
#pragma unroll
  for (int t = 0; t < 2; ++t) {
    const int rbase = row0 + t * 16 + hi * 8;
#pragma unroll
    for (int j = 0; j < 4; ++j) {
      const int col = n0 + j * 16 + l15;
      const float bv = bias[col];
#pragma unroll
      for (int r = 0; r < 8; ++r)
        Hb[(size_t)(rbase + r) * kD + col] = (__bf16)(acc[t * 4 + j][r] + bv);
    }
  }
}

// ---- GEMM2: logits = h @ W_attr^T + b_attr, plus per-column sum/sum^2 ------
// grid: (M/256), block: 256. N is a single padded 64-wide tile.

__global__ __launch_bounds__(256) void k_gemm_attr(
    const __bf16* __restrict__ Hb,    // [16384,768]
    const __bf16* __restrict__ Wab,   // [64,768] (rows >= 51 are zero)
    const float*  __restrict__ bias,  // [51]
    float* __restrict__ logits,       // [16384,64]
    float* __restrict__ gsum,         // [64]
    float* __restrict__ gsq) {        // [64]
  __shared__ __align__(16) __bf16 sB[2][64 * 64];
  __shared__ float s_sum[kAP];
  __shared__ float s_sq[kAP];
  if (threadIdx.x < kAP) { s_sum[threadIdx.x] = 0.f; s_sq[threadIdx.x] = 0.f; }

  const int tid  = threadIdx.x;
  const int lane = tid & 31;
  const int wave = tid >> 5;
  const int l15  = lane & 15;
  const int hi   = lane >> 4;
  const int row0 = blockIdx.x * 256 + wave * 32;

  const int crow = tid >> 3;
  const int csub = tid & 7;
  const __bf16* gB0 = Wab + (size_t)crow * kD + csub * 8;
  const __bf16* gB1 = gB0 + (size_t)32 * kD;
  const uint32_t ldsBase = (uint32_t)(uintptr_t)(&sB[0][0]);
  const uint32_t lds0 = ldsBase + (uint32_t)(crow * 128 + csub * 16);
  const uint32_t lds1 = lds0 + 32u * 128u;

  async_copy16(lds0, gB0);
  async_copy16(lds1, gB1);

  const __bf16* aB0 = Hb + (size_t)(row0 + l15) * kD + hi * 8;
  const __bf16* aB1 = aB0 + (size_t)16 * kD;

  v8f acc[8] = {v8f{}, v8f{}, v8f{}, v8f{}, v8f{}, v8f{}, v8f{}, v8f{}};

  for (int k0 = 0; k0 < kD; k0 += 64) {
    const int buf = (k0 >> 6) & 1;
    wait_async0();
    __syncthreads();
    if (k0 + 64 < kD) {
      const uint32_t bo = (uint32_t)((buf ^ 1) * 8192);
      async_copy16(lds0 + bo, gB0 + (k0 + 64));
      async_copy16(lds1 + bo, gB1 + (k0 + 64));
    }

    const __bf16* sb = &sB[buf][0];
#pragma unroll
    for (int s = 0; s < 2; ++s) {
      const int ke = k0 + s * 32;
      v16bf a0 = cat8(*(const v8bf*)(aB0 + ke), *(const v8bf*)(aB0 + ke + 16));
      v16bf a1 = cat8(*(const v8bf*)(aB1 + ke), *(const v8bf*)(aB1 + ke + 16));
      v16bf bfrag[4];
#pragma unroll
      for (int j = 0; j < 4; ++j) {
        const __bf16* bp = sb + (j * 16 + l15) * 64 + s * 32 + hi * 8;
        bfrag[j] = cat8(*(const v8bf*)bp, *(const v8bf*)(bp + 16));
      }
#pragma unroll
      for (int j = 0; j < 4; ++j) acc[j]     = wmma_bf16(a0, bfrag[j], acc[j]);
#pragma unroll
      for (int j = 0; j < 4; ++j) acc[4 + j] = wmma_bf16(a1, bfrag[j], acc[4 + j]);
    }
  }

#pragma unroll
  for (int t = 0; t < 2; ++t) {
    const int rbase = row0 + t * 16 + hi * 8;
#pragma unroll
    for (int j = 0; j < 4; ++j) {
      const int col = j * 16 + l15;          // same column for both half-waves
      const float bv = (col < kA) ? bias[col] : 0.f;
      float ls = 0.f, lq = 0.f;
#pragma unroll
      for (int r = 0; r < 8; ++r) {
        float v = acc[t * 4 + j][r] + bv;
        logits[(size_t)(rbase + r) * kAP + col] = v;
        ls += v;
        lq += v * v;
      }
      atomicAdd(&s_sum[col], ls);            // LDS ds_add_f32
      atomicAdd(&s_sq[col], lq);
    }
  }
  __syncthreads();
  if (threadIdx.x < kAP) {
    atomicAdd(&gsum[threadIdx.x], s_sum[threadIdx.x]);
    atomicAdd(&gsq[threadIdx.x], s_sq[threadIdx.x]);
  }
}

// ---- BatchNorm apply -------------------------------------------------------

__global__ void k_bn_apply(const float* __restrict__ logits,  // [16384,64]
                           const float* __restrict__ gsum,
                           const float* __restrict__ gsq,
                           const float* __restrict__ gamma,
                           const float* __restrict__ beta,
                           float* __restrict__ out) {          // [16384,51]
  int i = blockIdx.x * blockDim.x + threadIdx.x;
  if (i >= kB * kA) return;
  int row = i / kA;
  int col = i - row * kA;
  float invB = 1.f / (float)kB;
  float mean = gsum[col] * invB;
  float var  = gsq[col] * invB - mean * mean;   // biased, matches jnp.var
  float v = logits[(size_t)row * kAP + col];
  out[i] = (v - mean) * rsqrtf(var + kEps) * gamma[col] + beta[col];
}

// ---- launch ----------------------------------------------------------------

extern "C" void kernel_launch(void* const* d_in, const int* in_sizes, int n_in,
                              void* d_out, int out_size, void* d_ws, size_t ws_size,
                              hipStream_t stream) {
  const float* x     = (const float*)d_in[0];  // [16384,768]
  const float* We    = (const float*)d_in[1];  // [768,768]
  const float* be    = (const float*)d_in[2];  // [768]
  const float* Wa    = (const float*)d_in[3];  // [51,768]
  const float* ba    = (const float*)d_in[4];  // [51]
  const float* gamma = (const float*)d_in[5];  // [51]
  const float* beta  = (const float*)d_in[6];  // [51]
  float* out = (float*)d_out;

  char* ws = (char*)d_ws;
  size_t off = 0;
  auto take = [&](size_t bytes) {
    void* p = ws + off;
    off = (off + bytes + 255) & ~(size_t)255;
    return p;
  };
  __bf16* xb     = (__bf16*)take((size_t)kB * kD * 2);   // 24 MB
  __bf16* wb     = (__bf16*)take((size_t)kD * kD * 2);   // 1.125 MB
  __bf16* wab    = (__bf16*)take((size_t)kAP * kD * 2);  // 96 KB
  __bf16* hb     = (__bf16*)take((size_t)kB * kD * 2);   // 24 MB
  float*  logits = (float*)take((size_t)kB * kAP * 4);   // 4 MB
  float*  gsum   = (float*)take((size_t)kAP * 4);
  float*  gsq    = (float*)take((size_t)kAP * 4);

  // setup: bf16 conversion + padding + stats zeroing
  {
    int n = kB * kD;
    k_cvt_bf16<<<(n + 255) / 256, 256, 0, stream>>>(x, xb, n);
  }
  {
    int n = kD * kD;
    k_cvt_bf16<<<(n + 255) / 256, 256, 0, stream>>>(We, wb, n);
  }
  k_pad_wattr<<<(kAP * kD + 255) / 256, 256, 0, stream>>>(Wa, wab);
  k_zero_f32<<<1, 256, 0, stream>>>(gsum, kAP);
  k_zero_f32<<<1, 256, 0, stream>>>(gsq, kAP);

  // GEMM1: h = x @ We^T + be   -> bf16
  k_gemm_embed<<<dim3(kB / 256, kD / 64), 256, 0, stream>>>(xb, wb, be, hb);

  // GEMM2: logits = h @ Wa^T + ba, with column sum / sum^2 accumulation
  k_gemm_attr<<<kB / 256, 256, 0, stream>>>(hb, wab, ba, logits, gsum, gsq);

  // BatchNorm apply
  {
    int n = kB * kA;
    k_bn_apply<<<(n + 255) / 256, 256, 0, stream>>>(logits, gsum, gsq, gamma, beta, out);
  }
}
